// CKAN_10548439679187
// MI455X (gfx1250) — matrix operand order repaired and checked by reading, not compiled, not run on hardware
//
#include <hip/hip_runtime.h>
#include <hip/hip_bf16.h>
#include <cstddef>
#include <cstdint>

#define DIM 64
#define TT 64
#define NLAYER 2
#define BATCH 4096

typedef __attribute__((ext_vector_type(16))) _Float16 v16h;
typedef __attribute__((ext_vector_type(8)))  _Float16 v8h;
typedef __attribute__((ext_vector_type(4)))  _Float16 v4h;
typedef __attribute__((ext_vector_type(8)))  float    v8f;

union AFrag { v16h v; v8h h[2]; };

// A-fragment (16x32 f16, M = lane%16): elements 0..7 <- K=kbase..kbase+7,
// elements 8..15 <- K=kbase+16..kbase+23, where kbase = 32*kt + 8*(lane>=16).
__device__ __forceinline__ v16h load_a_frag(const _Float16* X, int ld, int m, int kbase) {
  AFrag u;
  u.h[0] = *(const v8h*)(X + m * ld + kbase);
  u.h[1] = *(const v8h*)(X + m * ld + kbase + 16);
  return u.v;
}

// ---------------------------------------------------------------------------
// Kernel 0: one-time (per launch) weight prep: f32 -> f16, transposed N-major
// so B-fragments are two contiguous 16B v8h loads per lane (as for A).
// ---------------------------------------------------------------------------
__global__ __launch_bounds__(128)
void ckan_prep(const float* __restrict__ w1, const float* __restrict__ w2,
               _Float16* __restrict__ W1t, _Float16* __restrict__ W2t) {
  const int stride = gridDim.x * blockDim.x;
  int tid = blockIdx.x * blockDim.x + threadIdx.x;
  for (int i = tid; i < 128 * DIM; i += stride) {
    int k = i >> 6, n = i & 63;
    W1t[n * 128 + k] = (_Float16)w1[i];
  }
  for (int i = tid; i < DIM * DIM; i += stride) {
    int k = i >> 6, n = i & 63;
    W2t[n * 64 + k] = (_Float16)w2[i];
  }
}

// ---------------------------------------------------------------------------
// Kernel 1: fused gather + attention MLP (WMMA) + softmax + weighted sum.
// One block per (b, layer, tower). 128 threads = 4 wave32; each wave owns a
// 16-row strip of the 64x64 GEMM outputs. t-tile gathered with
// global_load_async_to_lds_b128 (ASYNCcnt).
// ---------------------------------------------------------------------------
__global__ __launch_bounds__(128)
void ckan_att(const int* __restrict__ u_h, const int* __restrict__ u_r, const int* __restrict__ u_t,
              const int* __restrict__ i_h, const int* __restrict__ i_r, const int* __restrict__ i_t,
              const float* __restrict__ ent, const float* __restrict__ rel,
              const _Float16* __restrict__ W1t, const _Float16* __restrict__ W2t,
              const float* __restrict__ w3,
              float* __restrict__ ws_att) {
  __shared__ __align__(32) unsigned char smem[41744];
  _Float16* sXh = (_Float16*)(smem);           // [64][128] f16: concat(h,r) (phase 1)
  float*    sX2 = (float*)(smem);              // [64][64]  f32: relu(x@w2)  (phase 2, reuse)
  _Float16* sX1 = (_Float16*)(smem + 16384);   // [64][64]  f16: relu(x@w1)
  float*    sTm = (float*)(smem + 24576);      // [64][64]  f32: t tile (async-to-LDS)
  float*    sw3 = (float*)(smem + 40960);      // [64]
  float*    sA  = (float*)(smem + 41216);      // [64]
  float*    red = (float*)(smem + 41472);      // [64]
  float*    sS  = (float*)(smem + 41728);      // scalars
  __shared__ int sHi[TT], sRi[TT], sTi[TT];

  const int b = blockIdx.x, layer = blockIdx.y, tower = blockIdx.z;
  const int tid  = threadIdx.x;
  const int lane = tid & 31, wv = tid >> 5;
  const int hi   = lane >> 4, ln = lane & 15;

  const int* hI = (tower == 0 ? u_h : i_h) + ((size_t)layer * BATCH + b) * TT;
  const int* rI = (tower == 0 ? u_r : i_r) + ((size_t)layer * BATCH + b) * TT;
  const int* tI = (tower == 0 ? u_t : i_t) + ((size_t)layer * BATCH + b) * TT;

  if (tid < TT) {
    int hv = hI[tid], rv = rI[tid], tv = tI[tid];
    sHi[tid] = hv; sRi[tid] = rv; sTi[tid] = tv;
    __builtin_prefetch(ent + (size_t)hv * DIM, 0, 1);   // global_prefetch_b8
    __builtin_prefetch(ent + (size_t)tv * DIM, 0, 1);
    sw3[tid] = w3[tid];
  }
  __syncthreads();

  // h|r gather -> f16 LDS (float4 loads, packed v4h stores); 1024 16B chunks.
  for (int i = tid; i < TT * 16; i += 128) {
    int row = i >> 4, c4 = (i & 15) << 2;
    float4 hv = *(const float4*)(ent + (size_t)sHi[row] * DIM + c4);
    float4 rv = *(const float4*)(rel + (size_t)sRi[row] * DIM + c4);
    v4h hh = { (_Float16)hv.x, (_Float16)hv.y, (_Float16)hv.z, (_Float16)hv.w };
    v4h rr = { (_Float16)rv.x, (_Float16)rv.y, (_Float16)rv.z, (_Float16)rv.w };
    *(v4h*)(sXh + row * 128 + c4)      = hh;
    *(v4h*)(sXh + row * 128 + 64 + c4) = rr;
  }
  // t gather: raw f32 bytes, async DMA global -> LDS (no VGPR round trip).
  for (int i = tid; i < TT * 16; i += 128) {
    int row = i >> 4, c4 = (i & 15) << 2;
    int      goff = (int)(((unsigned)sTi[row] * DIM + (unsigned)c4) * sizeof(float));
    unsigned loff = (unsigned)(size_t)(sTm + row * 64 + c4);   // low 32b = LDS addr
    asm volatile("global_load_async_to_lds_b128 %0, %1, %2"
                 :: "v"(loff), "v"(goff), "s"(ent) : "memory");
  }
  asm volatile("s_wait_asynccnt 0" ::: "memory");
  __syncthreads();

  const int mrow = wv * 16 + ln;

  // GEMM1: X1 = relu([h|r] (64x128) @ W1 (128x64)); B-fragments straight from
  // global (L0-resident, identical across blocks).
  #pragma unroll
  for (int nt = 0; nt < 4; ++nt) {
    v8f acc = {};
    #pragma unroll
    for (int kt = 0; kt < 4; ++kt) {
      v16h a  = load_a_frag(sXh, 128, mrow, kt * 32 + hi * 8);
      v16h bm = load_a_frag(W1t, 128, nt * 16 + ln, kt * 32 + hi * 8);
      acc = __builtin_amdgcn_wmma_f32_16x16x32_f16(false, a, false, bm,
                                                   (short)0, acc, false, false);
    }
    #pragma unroll
    for (int v = 0; v < 8; ++v) {
      float x = acc[v];
      sX1[(wv * 16 + v + 8 * hi) * 64 + nt * 16 + ln] = (_Float16)(x > 0.f ? x : 0.f);
    }
  }
  __syncthreads();

  // GEMM2: X2 = relu(X1 (64x64) @ W2 (64x64)) -> f32 (reuses sXh region)
  #pragma unroll
  for (int nt = 0; nt < 4; ++nt) {
    v8f acc = {};
    #pragma unroll
    for (int kt = 0; kt < 2; ++kt) {
      v16h a  = load_a_frag(sX1, 64, mrow, kt * 32 + hi * 8);
      v16h bm = load_a_frag(W2t, 64, nt * 16 + ln, kt * 32 + hi * 8);
      acc = __builtin_amdgcn_wmma_f32_16x16x32_f16(false, a, false, bm,
                                                   (short)0, acc, false, false);
    }
    #pragma unroll
    for (int v = 0; v < 8; ++v) {
      float x = acc[v];
      sX2[(wv * 16 + v + 8 * hi) * 64 + nt * 16 + ln] = (x > 0.f ? x : 0.f);
    }
  }
  __syncthreads();

  // logits = X2 @ w3 ; sigmoid
  if (tid < TT) {
    float s = 0.f;
    #pragma unroll 8
    for (int n = 0; n < DIM; ++n) s += sX2[tid * 64 + n] * sw3[n];
    sA[tid] = 1.0f / (1.0f + expf(-s));
  }
  __syncthreads();

  // parallel softmax over T=64: max-reduce, exp, sum-reduce
  if (tid < TT) red[tid] = sA[tid];
  __syncthreads();
  #pragma unroll
  for (int off = 32; off > 0; off >>= 1) {
    if (tid < off) red[tid] = fmaxf(red[tid], red[tid + off]);
    __syncthreads();
  }
  float mx = red[0];
  __syncthreads();
  if (tid < TT) { float e = expf(sA[tid] - mx); sA[tid] = e; red[tid] = e; }
  __syncthreads();
  #pragma unroll
  for (int off = 32; off > 0; off >>= 1) {
    if (tid < off) red[tid] = red[tid] + red[tid + off];
    __syncthreads();
  }
  if (tid == 0) sS[0] = 1.0f / red[0];
  __syncthreads();

  // weighted sum over neighbors: out[d] = (1/Z) * sum_t exp_a[t] * t[t][d]
  if (tid < DIM) {
    float inv = sS[0];
    float o = 0.f;
    #pragma unroll 8
    for (int t = 0; t < TT; ++t) o += sA[t] * sTm[t * 64 + tid];
    ws_att[(((size_t)tower * NLAYER + layer) * BATCH + b) * DIM + tid] = o * inv;
  }
}

// ---------------------------------------------------------------------------
// Kernel 2: layer-0 head mean, cross-compress head, segment-wise dot, sigmoid
// ---------------------------------------------------------------------------
__device__ __forceinline__ float block_reduce64(float* red, float val) {
  const int d = threadIdx.x;
  red[d] = val;
  __syncthreads();
  #pragma unroll
  for (int off = 32; off > 0; off >>= 1) {
    if (d < off) red[d] = red[d] + red[d + off];
    __syncthreads();
  }
  float r = red[0];
  __syncthreads();
  return r;
}

__global__ __launch_bounds__(64)
void ckan_finalize(const int* __restrict__ users, const int* __restrict__ items,
                   const int* __restrict__ u_h, const int* __restrict__ i_h,
                   const float* __restrict__ ent,
                   const float* __restrict__ utab, const float* __restrict__ itab,
                   const float* __restrict__ ucc_wve, const float* __restrict__ ucc_wee,
                   const float* __restrict__ ucc_be,
                   const float* __restrict__ icc_wve, const float* __restrict__ icc_wee,
                   const float* __restrict__ icc_be,
                   const float* __restrict__ ws_att, float* __restrict__ out) {
  const int b = blockIdx.x, d = threadIdx.x;  // d in [0,64)
  __shared__ int   sidx[TT];
  __shared__ float red[TT];

  float kn[2], hd[2];
  for (int tw = 0; tw < 2; ++tw) {
    __syncthreads();
    const int* hI = (tw ? i_h : u_h) + (size_t)b * TT;  // layer 0 slice
    sidx[d] = hI[d];
    __syncthreads();
    float k = 0.f;
    for (int t = 0; t < TT; ++t) {
      if (t + 8 < TT) __builtin_prefetch(ent + (size_t)sidx[t + 8] * DIM + d, 0, 1);
      k += ent[(size_t)sidx[t] * DIM + d];
    }
    k *= (1.0f / TT);                                   // knowledge = h0.mean
    const int id = tw ? items[b] : users[b];
    const float v = (tw ? itab : utab)[(size_t)id * DIM + d];  // raw id emb
    const float* wve = tw ? icc_wve : ucc_wve;
    const float* wee = tw ? icc_wee : ucc_wee;
    const float* be  = tw ? icc_be  : ucc_be;
    // e_out = v * sum(e*wve) + e * sum(v*wee) + be
    float s_ve = block_reduce64(red, k * wve[d]);
    float s_ee = block_reduce64(red, v * wee[d]);
    kn[tw] = k;
    hd[tw] = v * s_ve + k * s_ee + be[d];
  }

  // dot(concat segs) = mean.mean + att0.att0 + att1.att1 + head.head
  float acc = kn[0] * kn[1] + hd[0] * hd[1];
  #pragma unroll
  for (int l = 0; l < NLAYER; ++l) {
    float au = ws_att[(((size_t)0 * NLAYER + l) * BATCH + b) * DIM + d];
    float av = ws_att[(((size_t)1 * NLAYER + l) * BATCH + b) * DIM + d];
    acc += au * av;
  }
  float tot = block_reduce64(red, acc);
  if (d == 0) out[b] = 1.0f / (1.0f + expf(-tot));
}

extern "C" void kernel_launch(void* const* d_in, const int* in_sizes, int n_in,
                              void* d_out, int out_size, void* d_ws, size_t ws_size,
                              hipStream_t stream) {
  (void)in_sizes; (void)n_in; (void)out_size; (void)ws_size;
  const int*   users = (const int*)d_in[0];
  const int*   items = (const int*)d_in[1];
  const int*   u_h   = (const int*)d_in[2];
  const int*   u_r   = (const int*)d_in[3];
  const int*   u_t   = (const int*)d_in[4];
  const int*   i_h   = (const int*)d_in[5];
  const int*   i_r   = (const int*)d_in[6];
  const int*   i_t   = (const int*)d_in[7];
  const float* ent   = (const float*)d_in[8];
  const float* rel   = (const float*)d_in[9];
  const float* utab  = (const float*)d_in[10];
  const float* itab  = (const float*)d_in[11];
  const float* w1    = (const float*)d_in[12];
  const float* w2    = (const float*)d_in[13];
  const float* w3    = (const float*)d_in[14];
  const float* ucc_wve = (const float*)d_in[17];
  const float* ucc_wee = (const float*)d_in[18];
  const float* ucc_be  = (const float*)d_in[20];
  const float* icc_wve = (const float*)d_in[23];
  const float* icc_wee = (const float*)d_in[24];
  const float* icc_be  = (const float*)d_in[26];
  float* out = (float*)d_out;

  // Workspace layout: [tower][layer][B][DIM] f32 att embeddings (4 MB),
  // then prepped f16 transposed weights (24 KB).
  float*     ws_att = (float*)d_ws;
  _Float16*  W1t    = (_Float16*)((char*)d_ws + (size_t)2 * NLAYER * BATCH * DIM * 4);
  _Float16*  W2t    = W1t + 128 * DIM;

  ckan_prep<<<32, 128, 0, stream>>>(w1, w2, W1t, W2t);

  dim3 g1(BATCH, NLAYER, 2);
  ckan_att<<<g1, 128, 0, stream>>>(u_h, u_r, u_t, i_h, i_r, i_t,
                                   ent, rel, W1t, W2t, w3, ws_att);
  ckan_finalize<<<BATCH, 64, 0, stream>>>(users, items, u_h, i_h, ent, utab, itab,
                                          ucc_wve, ucc_wee, ucc_be,
                                          icc_wve, icc_wee, icc_be,
                                          ws_att, out);
}
